// Seq2SeqAttention_57234734186838
// MI455X (gfx1250) — matrix operand order repaired and checked by reading, not compiled, not run on hardware
//
#include <hip/hip_runtime.h>

// ---------------------------------------------------------------------------
// Seq2Seq (GRU enc/dec + additive attention) for MI455X / gfx1250, wave32.
// All GEMMs via v_wmma_f32_16x16x32_bf16 (bf16 in, f32 accumulate).
// Key algorithmic hoist: enc_outs @ W_e^T precomputed once (attn_W split).
// ---------------------------------------------------------------------------

#define Bc   32
#define Sc   128
#define Tc   64
#define Ec   64
#define Hc   512
#define H3c  1536
#define Vc   32000
#define EHc  (Ec + Hc)   // 576

typedef __attribute__((ext_vector_type(16))) __bf16 v16bf;
typedef __attribute__((ext_vector_type(8)))  __bf16 v8bf;
typedef __attribute__((ext_vector_type(8)))  float  v8f;

// ---------------------------------------------------------------------------
// Generic bf16 GEMM:  C[M,N] = A[M,K] @ W[N,K]^T + bias[N]
// A row-major [M,K] bf16 (contiguous), W row-major [N,K] bf16, C f32 (ldc).
// 256 threads = 8 waves; wave -> one 16x16 tile; block covers 32 rows x 64 cols.
// M multiple of 32, N multiple of 64, K multiple of 32 (true for all calls).
// ---------------------------------------------------------------------------
__global__ __launch_bounds__(256)
void gemm_bf16_wmma(const __bf16* __restrict__ A, const __bf16* __restrict__ W,
                    const float* __restrict__ bias, float* __restrict__ C,
                    int M, int N, int K, int ldc)
{
    const int lane  = threadIdx.x & 31;
    const int wave  = threadIdx.x >> 5;
    const int mt    = wave & 1;              // 0..1
    const int nt    = wave >> 1;             // 0..3
    const int m0    = blockIdx.y * 32 + mt * 16;
    const int n0    = blockIdx.x * 64 + nt * 16;
    if (n0 >= N || m0 >= M) return;          // wave-uniform exit

    const int l16   = lane & 15;
    const int khalf = lane >> 4;             // 0 or 1

    // A fragment source: row m0+l16; two 8-elem contiguous K runs per step
    const __bf16* Arow = A + (size_t)(m0 + l16) * K;
    // B fragment source: column n0+l16 of W (row of row-major [N,K])
    const __bf16* Wrow = W + (size_t)(n0 + l16) * K;

    v8f acc = {};
    for (int k0 = 0; k0 < K; k0 += 32) {
        // prefetch the weight stream ~1KB ahead (speculative, safely dropped)
        __builtin_prefetch((const void*)(Wrow + k0 + 512), 0, 0);

        // A 16x32 bf16 layout (ISA 7.12.2): lanes 0-15 hold K=0..7 / 16..23,
        // lanes 16-31 hold K=8..15 / 24..31 for their M row.
        v8bf alo = *(const v8bf*)(Arow + k0 +      khalf * 8);
        v8bf ahi = *(const v8bf*)(Arow + k0 + 16 + khalf * 8);
        v16bf a = __builtin_shufflevector(alo, ahi,
            0,1,2,3,4,5,6,7, 8,9,10,11,12,13,14,15);

        // B 32x16 bf16 layout: lanes 0-15 hold K=0..7 / 16..23, lanes 16-31
        // K=8..15 / 24..31 for their N column (matches A).
        v8bf blo = *(const v8bf*)(Wrow + k0 +      khalf * 8);
        v8bf bhi = *(const v8bf*)(Wrow + k0 + 16 + khalf * 8);
        v16bf b = __builtin_shufflevector(blo, bhi,
            0,1,2,3,4,5,6,7, 8,9,10,11,12,13,14,15);

        acc = __builtin_amdgcn_wmma_f32_16x16x32_bf16(
            /*neg_a=*/false, a, /*neg_b=*/false, b,
            /*c_mod=*/(short)0, acc, /*reuse_a=*/false, /*reuse_b=*/false);
    }

    const int ncol = n0 + l16;
    const float bv = bias ? bias[ncol] : 0.0f;
    // C 16x16 f32 layout: VGPR r -> M = m0 + r + 8*khalf, N = ncol
    #pragma unroll
    for (int r = 0; r < 8; ++r) {
        const int m = m0 + khalf * 8 + r;
        C[(size_t)m * ldc + ncol] = acc[r] + bv;
    }
}

// ---------------------------------------------------------------------------
// Pointwise / data-movement helpers
// ---------------------------------------------------------------------------
__global__ void cvt_f32_bf16(const float* __restrict__ s, __bf16* __restrict__ d, int n)
{
    int i = blockIdx.x * blockDim.x + threadIdx.x;
    if (i < n) d[i] = (__bf16)s[i];
}

// dst[r*cols+c] = src[r*sstride + soff + c]  (used to split attn_W into W_h|W_e)
__global__ void cvt_f32_bf16_strided(const float* __restrict__ s, __bf16* __restrict__ d,
                                     int rows, int cols, int sstride, int soff)
{
    int i = blockIdx.x * blockDim.x + threadIdx.x;
    if (i >= rows * cols) return;
    int r = i / cols, c = i - r * cols;
    d[i] = (__bf16)s[(size_t)r * sstride + soff + c];
}

__global__ void zero_f32(float* __restrict__ d, int n)
{
    int i = blockIdx.x * blockDim.x + threadIdx.x;
    if (i < n) d[i] = 0.0f;
}

__global__ void zero_bf16(__bf16* __restrict__ d, int n)
{
    int i = blockIdx.x * blockDim.x + threadIdx.x;
    if (i < n) d[i] = (__bf16)0.0f;
}

// time-major embedding gather: dst[(t*B + b)*E + e] = table[tok[b*L + t]*E + e]
__global__ void gather_embed_bf16(const int* __restrict__ toks,
                                  const float* __restrict__ table,
                                  __bf16* __restrict__ dst, int L)
{
    int i = blockIdx.x * blockDim.x + threadIdx.x;
    if (i >= L * Bc * Ec) return;
    int e = i % Ec;
    int b = (i / Ec) % Bc;
    int t = i / (Ec * Bc);
    int tok = toks[b * L + t];
    dst[i] = (__bf16)table[(size_t)tok * Ec + e];
}

// copy one timestep of gathered token embeds into emb[:, 0:E)
__global__ void fill_emb_token(const __bf16* __restrict__ te, __bf16* __restrict__ emb)
{
    int i = blockIdx.x * blockDim.x + threadIdx.x;     // i < B*E
    if (i >= Bc * Ec) return;
    int b = i / Ec, e = i - b * Ec;
    emb[(size_t)b * EHc + e] = te[i];
}

// GRU pointwise combine. gi/gh: [B,3H] (bias already folded by GEMM).
// h' = (1-z)*tanh(i_n + r*h_n) + z*h ; writes f32 and bf16 copies.
__global__ void gru_combine(const float* __restrict__ gi, const float* __restrict__ gh,
                            const float* __restrict__ h,
                            float* __restrict__ hn, __bf16* __restrict__ hn_bf)
{
    int i = blockIdx.x * blockDim.x + threadIdx.x;     // i < B*H
    if (i >= Bc * Hc) return;
    int b = i / Hc, n = i - b * Hc;
    const float* gib = gi + (size_t)b * H3c;
    const float* ghb = gh + (size_t)b * H3c;
    float r  = 1.0f / (1.0f + expf(-(gib[n]          + ghb[n])));
    float z  = 1.0f / (1.0f + expf(-(gib[Hc + n]     + ghb[Hc + n])));
    float nn = tanhf(gib[2 * Hc + n] + r * ghb[2 * Hc + n]);
    float hv = (1.0f - z) * nn + z * h[i];
    hn[i]    = hv;
    hn_bf[i] = (__bf16)hv;
}

// ---------------------------------------------------------------------------
// Attention: scores(b,s) = v . tanh(hq[b] + eproj[s,b]) + v_b ; softmax over s;
// context(b) = sum_s w[s] * enc[s,b]; writes bf16 context into emb[:, E:E+H).
// One block per batch element, 8 waves.
// ---------------------------------------------------------------------------
__global__ __launch_bounds__(256)
void attention_kernel(const float* __restrict__ hq,     // [B,H]
                      const float* __restrict__ eproj,  // [S*B,H]
                      const float* __restrict__ enc,    // [S,B,H] f32
                      const float* __restrict__ vW,     // [H]
                      const float* __restrict__ vb,     // [1]
                      __bf16* __restrict__ emb)         // [B,E+H]
{
    const int b    = blockIdx.x;
    const int lane = threadIdx.x & 31;
    const int wave = threadIdx.x >> 5;
    __shared__ float sc[Sc];
    __shared__ float sinv;

    const float* hb = hq + (size_t)b * Hc;
    for (int s = wave; s < Sc; s += 8) {
        const float* ep = eproj + ((size_t)s * Bc + b) * Hc;
        float p = 0.0f;
        for (int k = lane; k < Hc; k += 32)
            p += vW[k] * tanhf(hb[k] + ep[k]);
        for (int off = 16; off > 0; off >>= 1)
            p += __shfl_xor(p, off, 32);
        if (lane == 0) sc[s] = p + vb[0];
    }
    __syncthreads();

    if (wave == 0) {
        float m = -3.4e38f;
        for (int s = lane; s < Sc; s += 32) m = fmaxf(m, sc[s]);
        for (int off = 16; off > 0; off >>= 1)
            m = fmaxf(m, __shfl_xor(m, off, 32));
        float sum = 0.0f;
        for (int s = lane; s < Sc; s += 32) {
            float e = expf(sc[s] - m);
            sc[s] = e;
            sum += e;
        }
        for (int off = 16; off > 0; off >>= 1)
            sum += __shfl_xor(sum, off, 32);
        if (lane == 0) sinv = 1.0f / sum;
    }
    __syncthreads();

    const float inv = sinv;
    for (int n = threadIdx.x; n < Hc; n += 256) {
        float acc = 0.0f;
        for (int s = 0; s < Sc; ++s)
            acc += sc[s] * enc[((size_t)s * Bc + b) * Hc + n];
        emb[(size_t)b * EHc + Ec + n] = (__bf16)(acc * inv);
    }
}

// ---------------------------------------------------------------------------
// Host orchestration (graph-capturable: only kernel launches + async d2d copies)
// ---------------------------------------------------------------------------
static inline dim3 gemm_grid(int M, int N) { return dim3(N / 64, M / 32); }

extern "C" void kernel_launch(void* const* d_in, const int* in_sizes, int n_in,
                              void* d_out, int out_size, void* d_ws, size_t ws_size,
                              hipStream_t stream)
{
    (void)in_sizes; (void)n_in; (void)out_size; (void)ws_size;

    const int*   src       = (const int*)  d_in[0];
    const int*   trg       = (const int*)  d_in[1];
    const float* src_embed = (const float*)d_in[2];
    const float* trg_embed = (const float*)d_in[3];
    const float* enc_W_ih  = (const float*)d_in[4];
    const float* enc_W_hh  = (const float*)d_in[5];
    const float* enc_b_ih  = (const float*)d_in[6];
    const float* enc_b_hh  = (const float*)d_in[7];
    const float* d0_W_ih   = (const float*)d_in[8];
    const float* d0_W_hh   = (const float*)d_in[9];
    const float* d0_b_ih   = (const float*)d_in[10];
    const float* d0_b_hh   = (const float*)d_in[11];
    const float* d1_W_ih   = (const float*)d_in[12];
    const float* d1_W_hh   = (const float*)d_in[13];
    const float* d1_b_ih   = (const float*)d_in[14];
    const float* d1_b_hh   = (const float*)d_in[15];
    const float* attn_W    = (const float*)d_in[16];
    const float* attn_b    = (const float*)d_in[17];
    const float* v_W       = (const float*)d_in[18];
    const float* v_b       = (const float*)d_in[19];
    const float* fc_W      = (const float*)d_in[20];
    const float* fc_b      = (const float*)d_in[21];
    float* out = (float*)d_out;

    // ---- workspace carve-out (256B aligned) ----
    char* ws = (char*)d_ws;
    size_t off = 0;
    auto alloc = [&](size_t bytes) -> char* {
        off = (off + 255) & ~(size_t)255;
        char* p = ws + off;
        off += bytes;
        return p;
    };
    __bf16* fcW_bf   = (__bf16*)alloc((size_t)Vc * Hc * 2);
    __bf16* eWih_bf  = (__bf16*)alloc((size_t)H3c * Ec * 2);
    __bf16* eWhh_bf  = (__bf16*)alloc((size_t)H3c * Hc * 2);
    __bf16* d0Wih_bf = (__bf16*)alloc((size_t)H3c * EHc * 2);
    __bf16* d0Whh_bf = (__bf16*)alloc((size_t)H3c * Hc * 2);
    __bf16* d1Wih_bf = (__bf16*)alloc((size_t)H3c * Hc * 2);
    __bf16* d1Whh_bf = (__bf16*)alloc((size_t)H3c * Hc * 2);
    __bf16* Wh_bf    = (__bf16*)alloc((size_t)Hc * Hc * 2);   // attn_W[:, :H]
    __bf16* We_bf    = (__bf16*)alloc((size_t)Hc * Hc * 2);   // attn_W[:, H:]
    __bf16* srcE_bf  = (__bf16*)alloc((size_t)Sc * Bc * Ec * 2);
    __bf16* trgE_bf  = (__bf16*)alloc((size_t)Tc * Bc * Ec * 2);
    float*  enc_f32  = (float*) alloc((size_t)Sc * Bc * Hc * 4);
    __bf16* enc_bf   = (__bf16*)alloc((size_t)Sc * Bc * Hc * 2);
    float*  eproj    = (float*) alloc((size_t)Sc * Bc * Hc * 4);
    float*  h0e_f32  = (float*) alloc((size_t)Bc * Hc * 4);   // zero initial h
    __bf16* h0e_bf   = (__bf16*)alloc((size_t)Bc * Hc * 2);
    float*  gi       = (float*) alloc((size_t)Bc * H3c * 4);
    float*  gh       = (float*) alloc((size_t)Bc * H3c * 4);
    float*  h0_f32   = (float*) alloc((size_t)Bc * Hc * 4);
    __bf16* h0_bf    = (__bf16*)alloc((size_t)Bc * Hc * 2);
    float*  h1_f32   = (float*) alloc((size_t)Bc * Hc * 4);
    __bf16* h1_bf    = (__bf16*)alloc((size_t)Bc * Hc * 2);
    float*  hq       = (float*) alloc((size_t)Bc * Hc * 4);
    __bf16* emb      = (__bf16*)alloc((size_t)Bc * EHc * 2);

    const int TPB = 256;
    auto blocks = [](int n) { return (n + 255) / 256; };

    // ---- one-time (per call) weight conversion + embedding gathers ----
    cvt_f32_bf16<<<blocks(Vc * Hc), TPB, 0, stream>>>(fc_W, fcW_bf, Vc * Hc);
    cvt_f32_bf16<<<blocks(H3c * Ec), TPB, 0, stream>>>(enc_W_ih, eWih_bf, H3c * Ec);
    cvt_f32_bf16<<<blocks(H3c * Hc), TPB, 0, stream>>>(enc_W_hh, eWhh_bf, H3c * Hc);
    cvt_f32_bf16<<<blocks(H3c * EHc), TPB, 0, stream>>>(d0_W_ih, d0Wih_bf, H3c * EHc);
    cvt_f32_bf16<<<blocks(H3c * Hc), TPB, 0, stream>>>(d0_W_hh, d0Whh_bf, H3c * Hc);
    cvt_f32_bf16<<<blocks(H3c * Hc), TPB, 0, stream>>>(d1_W_ih, d1Wih_bf, H3c * Hc);
    cvt_f32_bf16<<<blocks(H3c * Hc), TPB, 0, stream>>>(d1_W_hh, d1Whh_bf, H3c * Hc);
    cvt_f32_bf16_strided<<<blocks(Hc * Hc), TPB, 0, stream>>>(attn_W, Wh_bf, Hc, Hc, 2 * Hc, 0);
    cvt_f32_bf16_strided<<<blocks(Hc * Hc), TPB, 0, stream>>>(attn_W, We_bf, Hc, Hc, 2 * Hc, Hc);
    gather_embed_bf16<<<blocks(Sc * Bc * Ec), TPB, 0, stream>>>(src, src_embed, srcE_bf, Sc);
    gather_embed_bf16<<<blocks(Tc * Bc * Ec), TPB, 0, stream>>>(trg, trg_embed, trgE_bf, Tc);
    zero_f32 <<<blocks(Bc * Hc), TPB, 0, stream>>>(h0e_f32, Bc * Hc);
    zero_bf16<<<blocks(Bc * Hc), TPB, 0, stream>>>(h0e_bf, Bc * Hc);
    // outputs[:, 0, :] = 0
    zero_f32<<<blocks(Bc * Vc), TPB, 0, stream>>>(out, 1); // dummy-safe: real zero below
    {
        // zero the t=0 slice: out[b*T*V + v] for all b,v — strided, do with a grid
        // (reuse zero_f32 per-batch row)
        for (int b = 0; b < Bc; ++b)
            zero_f32<<<blocks(Vc), TPB, 0, stream>>>(out + (size_t)b * Tc * Vc, Vc);
    }

    // ---- encoder: 128 recurrent steps, all GEMMs on WMMA ----
    for (int t = 0; t < Sc; ++t) {
        const __bf16* xt = srcE_bf + (size_t)t * Bc * Ec;
        const __bf16* hp_bf  = (t == 0) ? h0e_bf  : enc_bf  + (size_t)(t - 1) * Bc * Hc;
        const float*  hp_f32 = (t == 0) ? h0e_f32 : enc_f32 + (size_t)(t - 1) * Bc * Hc;
        gemm_bf16_wmma<<<gemm_grid(Bc, H3c), TPB, 0, stream>>>(
            xt, eWih_bf, enc_b_ih, gi, Bc, H3c, Ec, H3c);
        gemm_bf16_wmma<<<gemm_grid(Bc, H3c), TPB, 0, stream>>>(
            hp_bf, eWhh_bf, enc_b_hh, gh, Bc, H3c, Hc, H3c);
        gru_combine<<<blocks(Bc * Hc), TPB, 0, stream>>>(
            gi, gh, hp_f32,
            enc_f32 + (size_t)t * Bc * Hc, enc_bf + (size_t)t * Bc * Hc);
    }

    // ---- hoisted attention precompute: eproj = enc_outs @ W_e^T + attn_b ----
    gemm_bf16_wmma<<<gemm_grid(Sc * Bc, Hc), TPB, 0, stream>>>(
        enc_bf, We_bf, attn_b, eproj, Sc * Bc, Hc, Hc, Hc);

    // ---- decoder initial hidden = encoder final hidden (both layers) ----
    hipMemcpyAsync(h0_f32, enc_f32 + (size_t)(Sc - 1) * Bc * Hc,
                   (size_t)Bc * Hc * 4, hipMemcpyDeviceToDevice, stream);
    hipMemcpyAsync(h1_f32, enc_f32 + (size_t)(Sc - 1) * Bc * Hc,
                   (size_t)Bc * Hc * 4, hipMemcpyDeviceToDevice, stream);
    hipMemcpyAsync(h0_bf, enc_bf + (size_t)(Sc - 1) * Bc * Hc,
                   (size_t)Bc * Hc * 2, hipMemcpyDeviceToDevice, stream);
    hipMemcpyAsync(h1_bf, enc_bf + (size_t)(Sc - 1) * Bc * Hc,
                   (size_t)Bc * Hc * 2, hipMemcpyDeviceToDevice, stream);

    // ---- decoder: 63 steps, logits written to out[:, i+1, :] ----
    for (int i = 0; i < Tc - 1; ++i) {
        // hq = h1 @ W_h^T   (the only per-step attention GEMM after hoisting)
        gemm_bf16_wmma<<<gemm_grid(Bc, Hc), TPB, 0, stream>>>(
            h1_bf, Wh_bf, (const float*)nullptr, hq, Bc, Hc, Hc, Hc);
        // emb[:, 0:E) = trg_embed[trg[:, i]]
        fill_emb_token<<<blocks(Bc * Ec), TPB, 0, stream>>>(
            trgE_bf + (size_t)i * Bc * Ec, emb);
        // softmax attention + context -> emb[:, E:E+H)
        attention_kernel<<<Bc, TPB, 0, stream>>>(hq, eproj, enc_f32, v_W, v_b, emb);
        // decoder layer 0 GRU
        gemm_bf16_wmma<<<gemm_grid(Bc, H3c), TPB, 0, stream>>>(
            emb, d0Wih_bf, d0_b_ih, gi, Bc, H3c, EHc, H3c);
        gemm_bf16_wmma<<<gemm_grid(Bc, H3c), TPB, 0, stream>>>(
            h0_bf, d0Whh_bf, d0_b_hh, gh, Bc, H3c, Hc, H3c);
        gru_combine<<<blocks(Bc * Hc), TPB, 0, stream>>>(gi, gh, h0_f32, h0_f32, h0_bf);
        // decoder layer 1 GRU
        gemm_bf16_wmma<<<gemm_grid(Bc, H3c), TPB, 0, stream>>>(
            h0_bf, d1Wih_bf, d1_b_ih, gi, Bc, H3c, Hc, H3c);
        gemm_bf16_wmma<<<gemm_grid(Bc, H3c), TPB, 0, stream>>>(
            h1_bf, d1Whh_bf, d1_b_hh, gh, Bc, H3c, Hc, H3c);
        gru_combine<<<blocks(Bc * Hc), TPB, 0, stream>>>(gi, gh, h1_f32, h1_f32, h1_bf);
        // logits = h1 @ fc_W^T + fc_b -> out[:, i+1, :]  (dominant GEMM, 32x32000x512)
        gemm_bf16_wmma<<<gemm_grid(Bc, Vc), TPB, 0, stream>>>(
            h1_bf, fcW_bf, fc_b, out + (size_t)(i + 1) * Vc, Bc, Vc, Hc, Tc * Vc);
    }
}
